// RuaAE_28183575396361
// MI455X (gfx1250) — compile-verified, matrix operand order checked by reading, not run on hardware
//
#include <hip/hip_runtime.h>
#include <hip/hip_bf16.h>
#include <math.h>

#define Bsz  8192
#define Lsz  224
#define Psz  6
#define Dsz  128
#define Ssz  25
#define CTRi 12
#define NPIX 8
#define LPAD 232              // bf16 row stride -> 464B rows, 16B aligned frag reads
#define NWG  (Bsz / NPIX)     // 1024

typedef __attribute__((ext_vector_type(16))) __bf16 v16bf;
typedef __attribute__((ext_vector_type(8)))  float  v8f;
typedef __attribute__((ext_vector_type(4)))  int    v4i;

union AFrag { v4i i[2]; v16bf v; };

__device__ __forceinline__ unsigned short f2bf(float f) {
  unsigned u = __float_as_uint(f);
  u += 0x7fffu + ((u >> 16) & 1u);          // round-to-nearest-even
  return (unsigned short)(u >> 16);
}
__device__ __forceinline__ float bf2f(unsigned short h) {
  return __uint_as_float(((unsigned)h) << 16);
}
__device__ __forceinline__ float sigm(float x) { return 1.f / (1.f + __expf(-x)); }

// LDS: Bw [128][LPAD] bf16 | Ax [NPIX][32][LPAD] bf16 | SCR [NPIX][1792] f32
#define SMEM_BYTES (Dsz*LPAD*2 + NPIX*32*LPAD*2 + NPIX*1792*4)

__global__ __launch_bounds__(256) void rua_front(
    const float* __restrict__ x,
    const float* __restrict__ wk_w, const float* __restrict__ wk_b,
    const float* __restrict__ wv_w, const float* __restrict__ wv_b,
    const float* __restrict__ wq_w, const float* __restrict__ wq_b,
    const float* __restrict__ ng_w, const float* __restrict__ ng_b,
    const float* __restrict__ f1_fw, const float* __restrict__ f1_fb,
    const float* __restrict__ f1_gw, const float* __restrict__ f1_gb,
    const float* __restrict__ s1_fw, const float* __restrict__ s1_fb,
    const float* __restrict__ s1_gw, const float* __restrict__ s1_gb,
    float* __restrict__ hpre, float* __restrict__ gpre,
    float* __restrict__ bn_part)
{
  extern __shared__ char smem[];
  unsigned short* Bw = (unsigned short*)smem;                 // [d][l]
  unsigned short* Ax = Bw + Dsz * LPAD;                       // [pix][s][l]
  float*          SCR = (float*)(Ax + NPIX * 32 * LPAD);

  const int tid  = threadIdx.x;
  const int wave = tid >> 5;
  const int lane = tid & 31;
  const int wg   = blockIdx.x;
  const int pix  = wg * NPIX + wave;

  // ---- stage wv (L,D) -> LDS bf16 as [d][l] (B-fragment friendly) ----
  for (int idx = tid; idx < Lsz * Dsz; idx += 256) {
    int l = idx / Dsz, d = idx % Dsz;                 // coalesced global read
    Bw[d * LPAD + l] = f2bf(wv_w[idx]);
  }
  // ---- stage this wave's pixel xt (25 x 224) -> LDS bf16 [s][l] ----
  const float* xp = x + (size_t)pix * (Lsz * Ssz);
  unsigned short* Ap = Ax + wave * 32 * LPAD;
  for (int e = lane; e < Lsz * Ssz; e += 32) {        // coalesced global read
    int l = e / Ssz, s = e % Ssz;
    Ap[s * LPAD + l] = f2bf(xp[e]);
  }
  for (int e = lane; e < 7 * LPAD; e += 32) Ap[25 * LPAD + e] = 0; // zero pad rows
  __syncthreads();

  float* q_   = SCR + wave * 1792;        // 128
  float* t_   = q_ + 128;                 // 224*4
  float* att_ = q_ + 1024;                // 4*32
  float* sur_ = q_ + 1152;                // 128
  float* cen_ = q_ + 1280;                // 128
  float* sw_  = q_ + 1408;                // 128
  float* stg_ = q_ + 1536;                // 128

  // ---- q[d] = xt[CENTER] . wq + wq_b ----
  {
    float qa[4];
#pragma unroll
    for (int j = 0; j < 4; ++j) qa[j] = wq_b[lane + 32 * j];
    const unsigned short* xc = Ap + CTRi * LPAD;
    for (int l = 0; l < Lsz; ++l) {
      float xv = bf2f(xc[l]);                         // LDS broadcast
      const float* wr = wq_w + l * Dsz + lane;
#pragma unroll
      for (int j = 0; j < 4; ++j) qa[j] = fmaf(xv, wr[32 * j], qa[j]);
    }
#pragma unroll
    for (int j = 0; j < 4; ++j) q_[lane + 32 * j] = qa[j];
  }

  // ---- t[l,h] = sum_{d in head h} wk[l,d]*q[d]  (keys-GEMM eliminated) ----
  for (int l = lane; l < Lsz; l += 32) {
    const float* wr = wk_w + l * Dsz;
    float a0 = 0, a1 = 0, a2 = 0, a3 = 0;
    for (int d = 0; d < 32; ++d) {
      a0 = fmaf(wr[d],      q_[d],      a0);
      a1 = fmaf(wr[32 + d], q_[32 + d], a1);
      a2 = fmaf(wr[64 + d], q_[64 + d], a2);
      a3 = fmaf(wr[96 + d], q_[96 + d], a3);
    }
    t_[l * 4 + 0] = a0; t_[l * 4 + 1] = a1; t_[l * 4 + 2] = a2; t_[l * 4 + 3] = a3;
  }

  // ---- bias term per head: c[h] = sum q[h]*wk_b[h] ----
  float ch[4];
#pragma unroll
  for (int h = 0; h < 4; ++h) {
    float c = 0.f;
    for (int d = 0; d < 32; ++d) c = fmaf(q_[h * 32 + d], wk_b[h * 32 + d], c);
    ch[h] = c;
  }

  // ---- scores + softmax (lane == s) ----
  float sc[4];
#pragma unroll
  for (int h = 0; h < 4; ++h) sc[h] = -1e30f;
  if (lane < Ssz) {
#pragma unroll
    for (int h = 0; h < 4; ++h) sc[h] = ch[h];
    const unsigned short* xr = Ap + lane * LPAD;
    for (int l = 0; l < Lsz; ++l) {
      float xv = bf2f(xr[l]);
#pragma unroll
      for (int h = 0; h < 4; ++h) sc[h] = fmaf(xv, t_[l * 4 + h], sc[h]);
    }
#pragma unroll
    for (int h = 0; h < 4; ++h) sc[h] *= 0.08838834764831845f;   // 1/sqrt(128)
    if (lane == CTRi) {
#pragma unroll
      for (int h = 0; h < 4; ++h) sc[h] -= 1e6f;                 // center mask
    }
  }
#pragma unroll
  for (int h = 0; h < 4; ++h) {
    float m = sc[h];
#pragma unroll
    for (int off = 16; off > 0; off >>= 1) m = fmaxf(m, __shfl_xor(m, off));
    float e = (lane < Ssz) ? __expf(sc[h] - m) : 0.f;
    float s = e;
#pragma unroll
    for (int off = 16; off > 0; off >>= 1) s += __shfl_xor(s, off);
    att_[h * 32 + lane] = (lane < Ssz) ? (e / s) : 0.f;
  }

  // ---- vals GEMM via WMMA, attention reduction fused in-register ----
  const int l15 = lane & 15;
  const int hi  = lane >> 4;
  const unsigned short* ArowA = Ap + l15 * LPAD;
  const unsigned short* ArowB = Ap + (16 + l15) * LPAD;
  for (int nt = 0; nt < 8; ++nt) {
    v8f a0 = {0, 0, 0, 0, 0, 0, 0, 0};
    v8f a1 = {0, 0, 0, 0, 0, 0, 0, 0};
    const unsigned short* brow = Bw + (nt * 16 + l15) * LPAD;
#pragma unroll
    for (int kk = 0; kk < 7; ++kk) {
      const int k0 = kk * 32 + hi * 8;
      AFrag fa0, fa1, fb;
      fa0.i[0] = *(const v4i*)(ArowA + k0);
      fa0.i[1] = *(const v4i*)(ArowA + k0 + 16);
      fa1.i[0] = *(const v4i*)(ArowB + k0);
      fa1.i[1] = *(const v4i*)(ArowB + k0 + 16);
      fb.i[0]  = *(const v4i*)(brow + k0);
      fb.i[1]  = *(const v4i*)(brow + k0 + 16);
      a0 = __builtin_amdgcn_wmma_f32_16x16x32_bf16(false, fa0.v, false, fb.v,
                                                   (short)0, a0, false, false);
      a1 = __builtin_amdgcn_wmma_f32_16x16x32_bf16(false, fa1.v, false, fb.v,
                                                   (short)0, a1, false, false);
    }
    const int h = nt >> 1;                            // head of this 16-col slab
    const float* ah = att_ + h * 32;
    float p = 0.f;
#pragma unroll
    for (int r = 0; r < 8; ++r) {                     // C layout: row = r + hi*8
      p += ah[hi * 8 + r]      * a0[r];
      p += ah[16 + hi * 8 + r] * a1[r];               // att[s>=25]==0
    }
    p += __shfl_xor(p, 16);                           // combine lane halves
    float cn = hi ? a0[4] : 0.f;                      // row s=12 lives at hi=1,r=4
    cn += __shfl_xor(cn, 16);
    const int n = nt * 16 + l15;
    if (hi == 0) {
      sur_[n] = p  + wv_b[n];                         // att sums to 1 -> +bias
      cen_[n] = cn + wv_b[n];
    }
  }

  // ---- nearby gate + ss + swish ----
  float dot = 0.f;
#pragma unroll
  for (int j = 0; j < 4; ++j) {
    int d = lane + 32 * j;
    dot += (cen_[d] - sur_[d]) * ng_w[d];
  }
#pragma unroll
  for (int off = 16; off > 0; off >>= 1) dot += __shfl_xor(dot, off);
  float nb = sigm(dot + ng_b[0]);
#pragma unroll
  for (int j = 0; j < 4; ++j) {
    int d = lane + 32 * j;
    float s_ = cen_[d] + nb * sur_[d];
    sw_[d] = s_ * sigm(s_);
  }

  // ---- GLU f1 & s1 (D->64), store hpre/gpre, stage BN partials ----
#pragma unroll
  for (int k = 0; k < 2; ++k) {
    int o = lane + 32 * k;
    float hf = f1_fb[o], hg = f1_gb[o], gf = s1_fb[o], gg = s1_gb[o];
    for (int d = 0; d < Dsz; ++d) {
      float z = sw_[d];
      hf = fmaf(z, f1_fw[d * 64 + o], hf);
      hg = fmaf(z, f1_gw[d * 64 + o], hg);
      gf = fmaf(z, s1_fw[d * 64 + o], gf);
      gg = fmaf(z, s1_gw[d * 64 + o], gg);
    }
    float hp = hf * sigm(hg);
    float gp = gf * sigm(gg);
    hpre[(size_t)pix * 64 + o] = hp;
    gpre[(size_t)pix * 64 + o] = gp;
    stg_[o] = hp; stg_[64 + o] = gp;
  }
  __syncthreads();
  {
    int o = tid & 63, grp = tid >> 6;   // 0:sum(h) 1:sum(h^2) 2:sum(g) 3:sum(g^2)
    float s = 0.f;
    for (int w = 0; w < NPIX; ++w) {
      float v = SCR[w * 1792 + 1536 + ((grp >= 2) ? 64 : 0) + o];
      s += (grp & 1) ? v * v : v;
    }
    bn_part[(size_t)wg * 256 + tid] = s;              // deterministic, no atomics
  }
}

__global__ __launch_bounds__(256) void rua_bnred(const float* __restrict__ bn_part,
                                                 float* __restrict__ bn_final) {
  int tid = threadIdx.x;
  float s = 0.f;
  for (int w = 0; w < NWG; ++w) s += bn_part[(size_t)w * 256 + tid];
  bn_final[tid] = s;
}

__global__ __launch_bounds__(256) void rua_back(
    const float* __restrict__ hpre, const float* __restrict__ gpre,
    const float* __restrict__ bn_final,
    const float* __restrict__ f_bn_g, const float* __restrict__ f_bn_b,
    const float* __restrict__ f2_fw, const float* __restrict__ f2_fb,
    const float* __restrict__ f2_gw, const float* __restrict__ f2_gb,
    const float* __restrict__ s_bn_g, const float* __restrict__ s_bn_b,
    const float* __restrict__ s2_fw, const float* __restrict__ s2_fb,
    const float* __restrict__ s2_gw, const float* __restrict__ s2_gb,
    const float* __restrict__ endmember,
    float* __restrict__ out_recon, float* __restrict__ out_abd,
    float* __restrict__ out_es)
{
  __shared__ float sh[NPIX][96];
  const int tid = threadIdx.x, wave = tid >> 5, lane = tid & 31;
  const int pix = blockIdx.x * NPIX + wave;
  const float invB = 1.f / (float)Bsz;

  // f path: BN + swish
#pragma unroll
  for (int k = 0; k < 2; ++k) {
    int o = lane + 32 * k;
    float m = bn_final[o] * invB;
    float v = bn_final[64 + o] * invB - m * m;
    float hn = (hpre[(size_t)pix * 64 + o] - m) * rsqrtf(v + 1e-5f) * f_bn_g[o] + f_bn_b[o];
    sh[wave][o] = hn * sigm(hn);
  }
  if (lane < Psz) {
    float f = f2_fb[lane], g = f2_gb[lane];
    for (int o = 0; o < 64; ++o) {
      float z = sh[wave][o];
      f = fmaf(z, f2_fw[o * Psz + lane], f);
      g = fmaf(z, f2_gw[o * Psz + lane], g);
    }
    sh[wave][64 + lane] = f * sigm(g);
  }
  float mx = -1e30f;
#pragma unroll
  for (int j = 0; j < Psz; ++j) mx = fmaxf(mx, sh[wave][64 + j]);
  float ex[Psz], sum = 0.f;
#pragma unroll
  for (int j = 0; j < Psz; ++j) { ex[j] = __expf(sh[wave][64 + j] - mx); sum += ex[j]; }
  float rinv = 1.f / sum;

  // s path: BN + swish (reuse sh[wave][0..63])
#pragma unroll
  for (int k = 0; k < 2; ++k) {
    int o = lane + 32 * k;
    float m = bn_final[128 + o] * invB;
    float v = bn_final[192 + o] * invB - m * m;
    float gn = (gpre[(size_t)pix * 64 + o] - m) * rsqrtf(v + 1e-5f) * s_bn_g[o] + s_bn_b[o];
    sh[wave][o] = gn * sigm(gn);
  }
  if (lane < Psz) {
    float f = s2_fb[lane], g = s2_gb[lane];
    for (int o = 0; o < 64; ++o) {
      float z = sh[wave][o];
      f = fmaf(z, s2_fw[o * Psz + lane], f);
      g = fmaf(z, s2_gw[o * Psz + lane], g);
    }
    sh[wave][70 + lane] = tanhf(f * sigm(g));
  }
  if (lane < Psz) {
    out_abd[(size_t)pix * Psz + lane] = ex[lane] * rinv;
    out_es [(size_t)pix * Psz + lane] = sh[wave][70 + lane];
  }
  float wj[Psz];
#pragma unroll
  for (int j = 0; j < Psz; ++j)
    wj[j] = (ex[j] * rinv) * (1.f + 0.2f * sh[wave][70 + j]);
  for (int l = lane; l < Lsz; l += 32) {
    float r = 0.f;
#pragma unroll
    for (int j = 0; j < Psz; ++j) r = fmaf(wj[j], endmember[j * Lsz + l], r);
    out_recon[(size_t)pix * Lsz + l] = r;
  }
}

extern "C" void kernel_launch(void* const* d_in, const int* in_sizes, int n_in,
                              void* d_out, int out_size, void* d_ws, size_t ws_size,
                              hipStream_t stream) {
  const float* x     = (const float*)d_in[0];
  const float* wk_w  = (const float*)d_in[1];
  const float* wk_b  = (const float*)d_in[2];
  const float* wv_w  = (const float*)d_in[3];
  const float* wv_b  = (const float*)d_in[4];
  const float* wq_w  = (const float*)d_in[5];
  const float* wq_b  = (const float*)d_in[6];
  const float* ng_w  = (const float*)d_in[7];
  const float* ng_b  = (const float*)d_in[8];
  const float* f1_fw = (const float*)d_in[9];
  const float* f1_fb = (const float*)d_in[10];
  const float* f1_gw = (const float*)d_in[11];
  const float* f1_gb = (const float*)d_in[12];
  const float* f_bn_g= (const float*)d_in[13];
  const float* f_bn_b= (const float*)d_in[14];
  const float* f2_fw = (const float*)d_in[15];
  const float* f2_fb = (const float*)d_in[16];
  const float* f2_gw = (const float*)d_in[17];
  const float* f2_gb = (const float*)d_in[18];
  const float* s1_fw = (const float*)d_in[19];
  const float* s1_fb = (const float*)d_in[20];
  const float* s1_gw = (const float*)d_in[21];
  const float* s1_gb = (const float*)d_in[22];
  const float* s_bn_g= (const float*)d_in[23];
  const float* s_bn_b= (const float*)d_in[24];
  const float* s2_fw = (const float*)d_in[25];
  const float* s2_fb = (const float*)d_in[26];
  const float* s2_gw = (const float*)d_in[27];
  const float* s2_gb = (const float*)d_in[28];
  const float* endm  = (const float*)d_in[29];

  float* ws = (float*)d_ws;
  float* hpre     = ws;
  float* gpre     = hpre + (size_t)Bsz * 64;
  float* bn_part  = gpre + (size_t)Bsz * 64;
  float* bn_final = bn_part + (size_t)NWG * 256;

  float* out_recon = (float*)d_out;
  float* out_abd   = out_recon + (size_t)Bsz * Lsz;
  float* out_es    = out_abd + (size_t)Bsz * Psz;

  rua_front<<<dim3(NWG), dim3(256), SMEM_BYTES, stream>>>(
      x, wk_w, wk_b, wv_w, wv_b, wq_w, wq_b, ng_w, ng_b,
      f1_fw, f1_fb, f1_gw, f1_gb, s1_fw, s1_fb, s1_gw, s1_gb,
      hpre, gpre, bn_part);
  rua_bnred<<<dim3(1), dim3(256), 0, stream>>>(bn_part, bn_final);
  rua_back<<<dim3(NWG), dim3(256), 0, stream>>>(
      hpre, gpre, bn_final, f_bn_g, f_bn_b, f2_fw, f2_fb, f2_gw, f2_gb,
      s_bn_g, s_bn_b, s2_fw, s2_fb, s2_gw, s2_gb, endm,
      out_recon, out_abd, out_es);
}
